// SlotAttentionSAVi_30640296689876
// MI455X (gfx1250) — matrix operand order repaired and checked by reading, not compiled, not run on hardware
//
#include <hip/hip_runtime.h>

#define B_    64
#define N_    4096
#define D_    256
#define KS_   11
#define H_    512
#define ITERS_ 5
#define SCALE_ 0.0625f     /* 256^-0.5 */
#define LN_EPS_ 1e-5f
#define EPS_    1e-8f

typedef __bf16 bf16;
typedef __attribute__((ext_vector_type(16))) __bf16 v16bf;
typedef __attribute__((ext_vector_type(8)))  __bf16 v8bf;
typedef __attribute__((ext_vector_type(8)))  float  v8f;

__device__ __forceinline__ bf16 f2bf(float f) {
  unsigned u = __builtin_bit_cast(unsigned, f);
  u += 0x7FFFu + ((u >> 16) & 1u);            // round to nearest even
  unsigned short h = (unsigned short)(u >> 16);
  return __builtin_bit_cast(bf16, h);
}

__device__ __forceinline__ v8f wmma_bf16(v16bf a, v16bf b, v8f c) {
  // D(f32 16x16) = A(bf16 16x32) * B(bf16 32x16) + C
  return __builtin_amdgcn_wmma_f32_16x16x32_bf16(false, a, false, b, (short)0, c,
                                                 false, false);
}

__device__ __forceinline__ v16bf cat8(v8bf lo, v8bf hi) {
  return __builtin_shufflevector(lo, hi, 0,1,2,3,4,5,6,7,8,9,10,11,12,13,14,15);
}

// A fragment (16x32, rows x K) from LDS tile stored row-major with STRIDE elems.
// Lane l<16: row l, K 0..15 ; lane l+16: row l, K 16..31.
template<int STRIDE>
__device__ __forceinline__ v16bf ldsA(const bf16* base, int rowBase, int kBase, int lane) {
  int row  = rowBase + (lane & 15);
  int half = lane >> 4;
  const bf16* p = base + row * STRIDE + kBase + half * 16;
  v8bf lo = *(const v8bf*)(p);
  v8bf hi = *(const v8bf*)(p + 8);
  return cat8(lo, hi);
}

// B fragment (32x16, K x cols) from a row-major [col][K] matrix with given
// row stride. Lane l<16: col l, K 0..15 ; lane l+16: col l, K 16..31 (32B).
__device__ __forceinline__ v16bf glbB(const bf16* base, size_t rowStride,
                                      int colBase, int kBase, int lane) {
  size_t col = (size_t)colBase + (lane & 15);
  int k = kBase + ((lane >> 4) << 4);
  return *(const v16bf*)(base + col * rowStride + k);
}

__device__ __forceinline__ float wave_sum(float v) {
  for (int m = 16; m >= 1; m >>= 1) v += __shfl_xor(v, m, 32);
  return v;
}

/* ---------------- Kernel 1: LayerNorm(inputs) -> x_bf16 ------------------- */
__global__ __launch_bounds__(256) void ln_x_kernel(const float* __restrict__ x,
                                                   const float* __restrict__ w,
                                                   const float* __restrict__ b,
                                                   bf16* __restrict__ out) {
  int wave = threadIdx.x >> 5, lane = threadIdx.x & 31;
  size_t row = (size_t)blockIdx.x * 8 + wave;           // 262144 rows total
  const float* p = x + row * D_ + lane * 8;
  float v[8]; float s = 0.f, sq = 0.f;
  #pragma unroll
  for (int j = 0; j < 8; ++j) { v[j] = p[j]; s += v[j]; sq += v[j] * v[j]; }
  s  = wave_sum(s);
  sq = wave_sum(sq);
  float m   = s * (1.f / D_);
  float var = sq * (1.f / D_) - m * m;
  float inv = rsqrtf(var + LN_EPS_);
  v8bf o;
  #pragma unroll
  for (int j = 0; j < 8; ++j) {
    int d = lane * 8 + j;
    o[j] = f2bf((v[j] - m) * inv * w[d] + b[d]);
  }
  *(v8bf*)(out + row * D_ + lane * 8) = o;
}

/* ---------------- Kernel 1b: Wk/Wv -> bf16 (once) ------------------------- */
__global__ void cvt_w_kernel(const float* __restrict__ Wk, const float* __restrict__ Wv,
                             bf16* __restrict__ Wk16, bf16* __restrict__ Wv16) {
  int idx = blockIdx.x * 256 + threadIdx.x;             // 65536 per matrix
  Wk16[idx] = f2bf(Wk[idx]);
  Wv16[idx] = f2bf(Wv[idx]);
}

/* -------- Kernel 2: k/v projection GEMM, bf16 WMMA, 128x64 block tile -----
 * z==0: k, stored row-major [b*N + n][d]   (B operand of dots)
 * z==1: v, stored TRANSPOSED [b*D + d][n]  (B operand of updates)           */
__global__ __launch_bounds__(256) void proj_gemm_kernel(const bf16* __restrict__ X,
                                                        const bf16* __restrict__ Wk16,
                                                        const bf16* __restrict__ Wv16,
                                                        bf16* __restrict__ K16,
                                                        bf16* __restrict__ VT16) {
  constexpr int AST = 48;                           // padded K-stride (96B rows)
  __shared__ bf16 As[128 * AST];
  const bf16* W = (blockIdx.z == 0) ? Wk16 : Wv16;
  size_t row0 = (size_t)blockIdx.x * 128;
  int    col0 = blockIdx.y * 64;
  int t = threadIdx.x, lane = t & 31, wv = t >> 5;
  int wr = wv >> 1, wc = wv & 1;                    // 4x2 wave grid, 32x32 each
  v8f acc[2][2];
  #pragma unroll
  for (int a = 0; a < 2; ++a)
    #pragma unroll
    for (int c = 0; c < 2; ++c)
      acc[a][c] = (v8f){0.f,0.f,0.f,0.f,0.f,0.f,0.f,0.f};

  for (int d0 = 0; d0 < D_; d0 += 32) {
    { // stage A: 128 rows x 32 K, 32B per thread
      int r = t >> 1, half = t & 1;
      const bf16* src = X + (row0 + r) * D_ + d0 + half * 16;
      if (d0 + 32 < D_) __builtin_prefetch(src + 32, 0, 0);
      *(v16bf*)&As[r * AST + half * 16] = *(const v16bf*)src;
    }
    __syncthreads();
    v16bf af[2], bfr[2];
    #pragma unroll
    for (int rt = 0; rt < 2; ++rt) af[rt]  = ldsA<AST>(As, wr * 32 + rt * 16, 0, lane);
    #pragma unroll
    for (int ct = 0; ct < 2; ++ct)                  // weights: L2-resident, direct
      bfr[ct] = glbB(W, D_, col0 + wc * 32 + ct * 16, d0, lane);
    #pragma unroll
    for (int rt = 0; rt < 2; ++rt)
      #pragma unroll
      for (int ct = 0; ct < 2; ++ct)
        acc[rt][ct] = wmma_bf16(af[rt], bfr[ct], acc[rt][ct]);
    __syncthreads();
  }
  // epilogue (C layout: lane owns col = lane&15, rows M = 8*(lane>=16)+i)
  int mh = (lane >> 4) * 8;
  if (blockIdx.z == 0) {
    #pragma unroll
    for (int rt = 0; rt < 2; ++rt) {
      size_t rb = row0 + wr * 32 + rt * 16;
      #pragma unroll
      for (int ct = 0; ct < 2; ++ct) {
        int col = col0 + wc * 32 + ct * 16 + (lane & 15);
        #pragma unroll
        for (int i = 0; i < 8; ++i)
          K16[(rb + mh + i) * D_ + col] = f2bf(acc[rt][ct][i]);
      }
    }
  } else {
    // v transposed: lane's 8 rows are consecutive n -> one 16B store each
    #pragma unroll
    for (int rt = 0; rt < 2; ++rt) {
      size_t rb = row0 + wr * 32 + rt * 16;           // global token row
      int bIdx = (int)(rb >> 12);                     // row / 4096
      int nIn  = (int)(rb & 4095) + mh;
      #pragma unroll
      for (int ct = 0; ct < 2; ++ct) {
        int d = col0 + wc * 32 + ct * 16 + (lane & 15);
        v8bf o;
        #pragma unroll
        for (int i = 0; i < 8; ++i) o[i] = f2bf(acc[rt][ct][i]);
        *(v8bf*)(VT16 + ((size_t)bIdx * D_ + d) * N_ + nIn) = o;
      }
    }
  }
}

/* -------- Kernel 3: q = LN(slots) @ Wq.T  (704 rows, VALU) ---------------- */
__global__ __launch_bounds__(256) void slots_q_kernel(const float* __restrict__ slots,
                                                      const float* __restrict__ lnw,
                                                      const float* __restrict__ lnb,
                                                      const float* __restrict__ Wq,
                                                      bf16* __restrict__ Q16) {
  __shared__ float pbuf[D_];
  __shared__ float red1[8], red2[8];
  int b = blockIdx.x, t = threadIdx.x, lane = t & 31, wv = t >> 5;
  for (int i = 0; i < KS_; ++i) {
    float x = slots[((size_t)b * KS_ + i) * D_ + t];
    float s1 = wave_sum(x), s2 = wave_sum(x * x);
    if (lane == 0) { red1[wv] = s1; red2[wv] = s2; }
    __syncthreads();
    float tot = 0.f, totq = 0.f;
    #pragma unroll
    for (int w = 0; w < 8; ++w) { tot += red1[w]; totq += red2[w]; }
    float m   = tot * (1.f / D_);
    float var = totq * (1.f / D_) - m * m;
    pbuf[t] = (x - m) * rsqrtf(var + LN_EPS_) * lnw[t] + lnb[t];
    __syncthreads();
    float acc = 0.f;
    const float* wr = Wq + (size_t)t * D_;
    for (int d = 0; d < D_; ++d) acc += pbuf[d] * wr[d];
    Q16[((size_t)b * 16 + i) * D_ + t] = f2bf(acc);
    __syncthreads();
  }
  for (int i = KS_; i < 16; ++i)            // zero slot-padding rows 11..15
    Q16[((size_t)b * 16 + i) * D_ + t] = f2bf(0.f);
}

/* -------- Kernel 4: zero U/S accumulators --------------------------------- */
__global__ void zero_us_kernel(float* __restrict__ U, float* __restrict__ S) {
  int idx = blockIdx.x * 256 + threadIdx.x;
  if (idx < B_ * 16 * D_) U[idx] = 0.f;
  if (idx < B_ * 16)      S[idx] = 0.f;
}

/* -------- Kernel 5: fused dots -> softmax(slots) -> updates --------------- */
__global__ __launch_bounds__(256) void attn_kernel(const bf16* __restrict__ Q16,
                                                   const bf16* __restrict__ K16,
                                                   const bf16* __restrict__ VT16,
                                                   float* __restrict__ U,
                                                   float* __restrict__ S) {
  constexpr int QST = 272, AT = 272;
  __shared__ bf16 qs[16 * QST];        // q: 16 slot rows x 256 d
  __shared__ bf16 attnS[16 * AT];      // attn: 16 slot rows x 256 n (A layout)
  __shared__ float Sloc[16];
  int b = blockIdx.y, chunk = blockIdx.x;        // chunk of 256 columns of N
  int t = threadIdx.x, lane = t & 31, wv = t >> 5;
  size_t bRow = (size_t)b * N_;

  { // stage q (8KB)
    int slot = t >> 4, seg = t & 15;
    *(v16bf*)&qs[slot * QST + seg * 16] =
        *(const v16bf*)(Q16 + ((size_t)b * 16 + slot) * D_ + seg * 16);
  }
  if (t < 16) Sloc[t] = 0.f;
  __syncthreads();

  /* phase 1: dots + masked softmax over slots; each wave does 2 col-tiles */
  const bf16* kBase = K16 + bRow * D_;
  int rbase = (lane >> 4) * 8;                   // rows owned: rbase..rbase+7
  #pragma unroll
  for (int tsel = 0; tsel < 2; ++tsel) {
    int tt = wv + tsel * 8;
    int ntile = chunk * 256 + tt * 16;
    v8f acc = (v8f){0.f,0.f,0.f,0.f,0.f,0.f,0.f,0.f};
    for (int d0 = 0; d0 < D_; d0 += 32) {
      v16bf a   = ldsA<QST>(qs, 0, d0, lane);
      v16bf bfr = glbB(kBase, D_, ntile, d0, lane);
      acc = wmma_bf16(a, bfr, acc);
    }
    float dv[8], ev[8];
    float m = -3.0e38f;
    #pragma unroll
    for (int i = 0; i < 8; ++i) {
      dv[i] = (rbase + i < KS_) ? acc[i] * SCALE_ : -3.0e38f;
      m = fmaxf(m, dv[i]);
    }
    m = fmaxf(m, __shfl_xor(m, 16, 32));         // partner lane = other 8 rows
    float ssum = 0.f;
    #pragma unroll
    for (int i = 0; i < 8; ++i) {
      ev[i] = (rbase + i < KS_) ? __expf(dv[i] - m) : 0.f;
      ssum += ev[i];
    }
    ssum += __shfl_xor(ssum, 16, 32);
    float inv = 1.f / ssum;
    int colL = tt * 16 + (lane & 15);
    #pragma unroll
    for (int i = 0; i < 8; ++i) {
      float a_ = (rbase + i < KS_) ? ev[i] * inv + EPS_ : 0.f;
      attnS[(rbase + i) * AT + colL] = f2bf(a_);
      if (rbase + i < KS_) atomicAdd(&Sloc[rbase + i], a_);
    }
  }
  __syncthreads();
  if (t < 16) atomicAdd(&S[b * 16 + t], Sloc[t]);

  /* phase 2: U_partial = attn(16 x 256n) @ v(256n x 256d); wave owns 32 d.
   * v read directly from the transposed global layout vT[b][d][n].         */
  const bf16* vtBase = VT16 + (size_t)b * D_ * N_;
  v8f acc0 = (v8f){0.f,0.f,0.f,0.f,0.f,0.f,0.f,0.f};
  v8f acc1 = acc0;
  int dcol = wv * 32;
  for (int ns = 0; ns < 8; ++ns) {
    int nk = chunk * 256 + ns * 32;
    v16bf a  = ldsA<AT>(attnS, 0, ns * 32, lane);
    v16bf b0 = glbB(vtBase, N_, dcol,      nk, lane);
    v16bf b1 = glbB(vtBase, N_, dcol + 16, nk, lane);
    acc0 = wmma_bf16(a, b0, acc0);
    acc1 = wmma_bf16(a, b1, acc1);
  }
  int c0 = dcol + (lane & 15);
  #pragma unroll
  for (int i = 0; i < 8; ++i) {
    atomicAdd(&U[((size_t)b * 16 + rbase + i) * D_ + c0],      acc0[i]);
    atomicAdd(&U[((size_t)b * 16 + rbase + i) * D_ + c0 + 16], acc1[i]);
  }
}

/* -------- Kernel 6: GRU cell + residual MLP (per (b,slot) workgroup) ------ */
__global__ __launch_bounds__(256) void gru_mlp_kernel(const float* __restrict__ U,
                                                      const float* __restrict__ S,
                                                      const float* __restrict__ slots,
                                                      const float* __restrict__ W_ih,
                                                      const float* __restrict__ b_ih,
                                                      const float* __restrict__ W_hh,
                                                      const float* __restrict__ b_hh,
                                                      const float* __restrict__ lnw,
                                                      const float* __restrict__ lnb,
                                                      const float* __restrict__ W1,
                                                      const float* __restrict__ b1,
                                                      const float* __restrict__ W2,
                                                      const float* __restrict__ b2,
                                                      float* __restrict__ dst) {
  __shared__ float upd[D_], hprev[D_], pbuf[D_], hm[H_];
  __shared__ float red1[8], red2[8];
  int i = blockIdx.x, b = blockIdx.y;
  int t = threadIdx.x, lane = t & 31, wv = t >> 5;
  float Sv = S[b * 16 + i];
  upd[t]   = U[((size_t)b * 16 + i) * D_ + t] / Sv;
  hprev[t] = slots[((size_t)b * KS_ + i) * D_ + t];
  __syncthreads();
  float gir = b_ih[t], giz = b_ih[D_ + t], gin = b_ih[2 * D_ + t];
  float ghr = b_hh[t], ghz = b_hh[D_ + t], ghn = b_hh[2 * D_ + t];
  const float* wi0 = W_ih + (size_t)t * D_;
  const float* wh0 = W_hh + (size_t)t * D_;
  for (int d = 0; d < D_; ++d) {
    float u = upd[d], h = hprev[d];
    gir += u * wi0[d];
    giz += u * wi0[(size_t)D_ * D_ + d];
    gin += u * wi0[(size_t)2 * D_ * D_ + d];
    ghr += h * wh0[d];
    ghz += h * wh0[(size_t)D_ * D_ + d];
    ghn += h * wh0[(size_t)2 * D_ * D_ + d];
  }
  float r  = 1.f / (1.f + __expf(-(gir + ghr)));
  float z  = 1.f / (1.f + __expf(-(giz + ghz)));
  float nn = tanhf(gin + r * ghn);
  float hn = (1.f - z) * nn + z * hprev[t];
  // LayerNorm(hn row)
  float s1 = wave_sum(hn), s2 = wave_sum(hn * hn);
  if (lane == 0) { red1[wv] = s1; red2[wv] = s2; }
  __syncthreads();
  float tot = 0.f, totq = 0.f;
  #pragma unroll
  for (int w = 0; w < 8; ++w) { tot += red1[w]; totq += red2[w]; }
  float m   = tot * (1.f / D_);
  float var = totq * (1.f / D_) - m * m;
  pbuf[t] = (hn - m) * rsqrtf(var + LN_EPS_) * lnw[t] + lnb[t];
  __syncthreads();
  #pragma unroll
  for (int rep = 0; rep < 2; ++rep) {
    int o = t + rep * D_;
    float acc = b1[o];
    const float* wr = W1 + (size_t)o * D_;
    for (int d = 0; d < D_; ++d) acc += pbuf[d] * wr[d];
    hm[o] = fmaxf(acc, 0.f);
  }
  __syncthreads();
  float acc = b2[t];
  const float* wr2 = W2 + (size_t)t * H_;
  for (int d = 0; d < H_; ++d) acc += hm[d] * wr2[d];
  dst[((size_t)b * KS_ + i) * D_ + t] = hn + acc;
}

/* -------- Kernel 7: slots_init -> slots workspace ------------------------- */
__global__ void copy_slots_kernel(const float* __restrict__ src, float* __restrict__ dst) {
  size_t idx = (size_t)blockIdx.x * 256 + threadIdx.x;
  dst[idx] = src[idx];
}

extern "C" void kernel_launch(void* const* d_in, const int* in_sizes, int n_in,
                              void* d_out, int out_size, void* d_ws, size_t ws_size,
                              hipStream_t stream) {
  const float* inputs     = (const float*)d_in[0];
  const float* slots_init = (const float*)d_in[1];
  const float* ln_f_w = (const float*)d_in[2];
  const float* ln_f_b = (const float*)d_in[3];
  const float* ln_s_w = (const float*)d_in[4];
  const float* ln_s_b = (const float*)d_in[5];
  const float* ln_p_w = (const float*)d_in[6];
  const float* ln_p_b = (const float*)d_in[7];
  const float* Wq   = (const float*)d_in[8];
  const float* Wk   = (const float*)d_in[9];
  const float* Wv   = (const float*)d_in[10];
  const float* W_ih = (const float*)d_in[11];
  const float* b_ih = (const float*)d_in[12];
  const float* W_hh = (const float*)d_in[13];
  const float* b_hh = (const float*)d_in[14];
  const float* W1   = (const float*)d_in[15];
  const float* b1   = (const float*)d_in[16];
  const float* W2   = (const float*)d_in[17];
  const float* b2   = (const float*)d_in[18];
  float* out = (float*)d_out;

  // workspace carve-up
  char* ws = (char*)d_ws;
  const size_t xkvBytes = (size_t)B_ * N_ * D_ * 2;     // 128 MB each
  bf16*  X16  = (bf16*)ws;              ws += xkvBytes;
  bf16*  K16  = (bf16*)ws;              ws += xkvBytes;
  bf16*  VT16 = (bf16*)ws;              ws += xkvBytes;
  bf16*  Q16  = (bf16*)ws;              ws += (size_t)B_ * 16 * D_ * 2;
  float* U    = (float*)ws;             ws += (size_t)B_ * 16 * D_ * 4;
  float* S    = (float*)ws;             ws += (size_t)B_ * 16 * 4;
  float* slots = (float*)ws;            ws += (size_t)B_ * KS_ * D_ * 4;
  bf16*  Wk16 = (bf16*)ws;              ws += (size_t)D_ * D_ * 2;
  bf16*  Wv16 = (bf16*)ws;              ws += (size_t)D_ * D_ * 2;

  // 1) LN over inputs -> bf16 ; weights -> bf16 (once)
  ln_x_kernel<<<(B_ * N_) / 8, 256, 0, stream>>>(inputs, ln_f_w, ln_f_b, X16);
  cvt_w_kernel<<<(D_ * D_) / 256, 256, 0, stream>>>(Wk, Wv, Wk16, Wv16);
  // 2) k (row-major) and v (transposed) projections via bf16 WMMA
  proj_gemm_kernel<<<dim3((B_ * N_) / 128, D_ / 64, 2), 256, 0, stream>>>(
      X16, Wk16, Wv16, K16, VT16);
  // 3) slots <- slots_init
  copy_slots_kernel<<<(B_ * KS_ * D_) / 256, 256, 0, stream>>>(slots_init, slots);

  for (int it = 0; it < ITERS_; ++it) {
    slots_q_kernel<<<B_, 256, 0, stream>>>(slots, ln_s_w, ln_s_b, Wq, Q16);
    zero_us_kernel<<<(B_ * 16 * D_) / 256, 256, 0, stream>>>(U, S);
    attn_kernel<<<dim3(N_ / 256, B_), 256, 0, stream>>>(Q16, K16, VT16, U, S);
    float* dst = (it == ITERS_ - 1) ? out : slots;
    gru_mlp_kernel<<<dim3(KS_, B_), 256, 0, stream>>>(
        U, S, slots, W_ih, b_ih, W_hh, b_hh, ln_p_w, ln_p_b, W1, b1, W2, b2, dst);
  }
}
